// AttentionBlock_25159918420592
// MI455X (gfx1250) — matrix operand order repaired and checked by reading, not compiled, not run on hardware
//
#include <hip/hip_runtime.h>

// ---------------------------------------------------------------------------
// Sigmoid-attention block for MI455X (gfx1250, wave32, WMMA).
//   q,k,v = Linear(x)            (8192x256 @ 256x256 + bias)
//   out   = sigmoid(q @ k^T) @ v -- streamed, attn matrix never hits HBM
// Matmuls: v_wmma_f32_16x16x32_f16 (f16 in, f32 accumulate).
// Staging: global_load_async_to_lds_b128, double-buffered on ASYNCcnt.
// ---------------------------------------------------------------------------

#define N_TOK 8192
#define HID   256

typedef __attribute__((ext_vector_type(16))) _Float16 v16h;
typedef __attribute__((ext_vector_type(8)))  _Float16 v8h;
typedef __attribute__((ext_vector_type(8)))  float    v8f;

__device__ __forceinline__ v8f wmma_f16(v16h a, v16h b, v8f c) {
  return __builtin_amdgcn_wmma_f32_16x16x32_f16(
      false, a, false, b, (short)0, c, /*reuse_a=*/false, /*reuse_b=*/false);
}
// Same-shape WMMA with A-operand reuse hint: legal because it is issued
// back-to-back after an identical instruction reading the same A registers.
__device__ __forceinline__ v8f wmma_f16_ra(v16h a, v16h b, v8f c) {
  return __builtin_amdgcn_wmma_f32_16x16x32_f16(
      false, a, false, b, (short)0, c, /*reuse_a=*/true, /*reuse_b=*/false);
}

// Flat shared-aperture address: low 32 bits are the LDS offset (ISA 10.2).
__device__ __forceinline__ unsigned lds_off(const void* p) {
  return (unsigned)(uintptr_t)p;
}
// Async DMA: 16 bytes global -> LDS, tracked on ASYNCcnt (ISA 10.x / 15.18).
__device__ __forceinline__ void async_ld_b128(unsigned ldsaddr, const void* gaddr) {
  asm volatile("global_load_async_to_lds_b128 %0, %1, off"
               :: "v"(ldsaddr), "v"(gaddr) : "memory");
}
__device__ __forceinline__ void wait_asynccnt0() {
  asm volatile("s_wait_asynccnt 0x0" ::: "memory");
}

// ---------------------------------------------------------------------------
// Phase 1: out = X @ W^T + b, emitted as f16.
// transposed==0 : out[row][col] (Q, K);  ==1 : out[col][row] (V -> Vt).
// ---------------------------------------------------------------------------
__global__ __launch_bounds__(128) void proj_kernel(
    const float* __restrict__ X, const float* __restrict__ W,
    const float* __restrict__ bias, _Float16* __restrict__ out, int transposed)
{
  __shared__ __align__(32) _Float16 ldsW[HID * HID];   // 128 KB

  const int tid  = threadIdx.x;
  const int lane = tid & 31;
  const int wave = tid >> 5;
  const int n    = lane & 15;
  const int g    = lane >> 4;
  const int row0 = (blockIdx.x * 4 + wave) * 16;

  for (int i = 0; i < 64; ++i) {               // W fp32 -> f16 into LDS
    int idx = tid + 128 * i;
    const float* s = W + 8 * idx;
    v8h w;
#pragma unroll
    for (int j = 0; j < 8; ++j) w[j] = (_Float16)s[j];
    *(v8h*)(ldsW + 8 * idx) = w;
  }
  __syncthreads();

  v16h afrag[8];                               // A rows row0..row0+15
  const float* xrow = X + (size_t)(row0 + n) * HID;
#pragma unroll
  for (int h = 0; h < 8; ++h) {
    const float* p0 = xrow + 32 * h + 16 * g;
    const float* p1 = p0 + 8;
    v16h a;
#pragma unroll
    for (int i = 0; i < 8; ++i) { a[i] = (_Float16)p0[i]; a[8 + i] = (_Float16)p1[i]; }
    afrag[h] = a;
  }

#pragma unroll
  for (int c = 0; c < 16; ++c) {
    v8f acc = {};
#pragma unroll
    for (int h = 0; h < 8; ++h) {
      v16h b = *(const v16h*)(ldsW + (16 * c + n) * HID + 32 * h + 16 * g);
      acc = wmma_f16(afrag[h], b, acc);
    }
    const float bv = bias[16 * c + n];
    if (!transposed) {
#pragma unroll
      for (int r = 0; r < 8; ++r)
        out[(size_t)(row0 + 8 * g + r) * HID + 16 * c + n] = (_Float16)(acc[r] + bv);
    } else {
      v8h o;
#pragma unroll
      for (int r = 0; r < 8; ++r) o[r] = (_Float16)(acc[r] + bv);
      *(v8h*)(out + (size_t)(16 * c + n) * N_TOK + row0 + 8 * g) = o;
    }
  }
}

// ---------------------------------------------------------------------------
// Phase 2: out = sigmoid(Qh @ Kh^T) @ V, kv streamed in tiles of 32 with
// double-buffered async global->LDS staging (one barrier per tile).
// ---------------------------------------------------------------------------
__global__ __launch_bounds__(128) void attn_kernel(
    const _Float16* __restrict__ Qh, const _Float16* __restrict__ Kh,
    const _Float16* __restrict__ Vt, float* __restrict__ out)
{
  __shared__ __align__(32) _Float16 ldsK[2][32 * HID];   // [buf][kvrow][hid] 32 KB
  __shared__ __align__(32) _Float16 ldsV[2][HID * 32];   // [buf][hid][kvrow] 32 KB
  __shared__ __align__(32) _Float16 ldsP[4][16 * 32];    // per-wave P scratch 4 KB

  const int tid  = threadIdx.x;
  const int lane = tid & 31;
  const int wave = tid >> 5;
  const int n    = lane & 15;
  const int g    = lane >> 4;
  const int qrow0 = (blockIdx.x * 4 + wave) * 16;

  // Q A-fragments resident in registers (64 VGPRs).
  v16h qfrag[8];
  const _Float16* qrow = Qh + (size_t)(qrow0 + n) * HID;
#pragma unroll
  for (int h = 0; h < 8; ++h) {
    v8h lo = *(const v8h*)(qrow + 32 * h + 16 * g);
    v8h hi = *(const v8h*)(qrow + 32 * h + 16 * g + 8);
    v16h a;
#pragma unroll
    for (int i = 0; i < 8; ++i) { a[i] = lo[i]; a[8 + i] = hi[i]; }
    qfrag[h] = a;
  }

  // Async stage of one 32-row kv tile (K: 16 KB flat, Vt: 256 x 64 B slices).
  auto stage = [&](int buf, int kv0) {
    const _Float16* srcK = Kh + (size_t)kv0 * HID;
    unsigned dK = lds_off(&ldsK[buf][0]);
    unsigned dV = lds_off(&ldsV[buf][0]);
#pragma unroll
    for (int i = 0; i < 8; ++i) {
      int idx = tid + 128 * i;                         // b128 index
      async_ld_b128(dK + 16 * idx, srcK + 8 * idx);
    }
#pragma unroll
    for (int i = 0; i < 8; ++i) {
      int idx = tid + 128 * i;
      int hid = idx >> 2, part = idx & 3;
      async_ld_b128(dV + 16 * idx, Vt + (size_t)hid * N_TOK + kv0 + 8 * part);
    }
  };

  v8f oacc[16] = {};

  const int T = N_TOK / 32;
  stage(0, 0);
  wait_asynccnt0();
  __syncthreads();

  for (int t = 0; t < T; ++t) {
    const int buf = t & 1;
    if (t + 1 < T) stage(buf ^ 1, (t + 1) * 32);       // prefetch next tile

    // S = Q @ K^T (16 x 32): two 16x16 tiles, K=256 in 8 WMMA steps each.
#pragma unroll
    for (int jt = 0; jt < 2; ++jt) {
      v8f s = {};
#pragma unroll
      for (int h = 0; h < 8; ++h) {
        v16h b = *(const v16h*)(&ldsK[buf][(16 * jt + n) * HID + 32 * h + 16 * g]);
        s = wmma_f16(qfrag[h], b, s);
      }
#pragma unroll
      for (int r = 0; r < 8; ++r) {                    // sigmoid -> f16 -> LDS
        float p = 1.0f / (1.0f + __expf(-s[r]));
        ldsP[wave][(8 * g + r) * 32 + 16 * jt + n] = (_Float16)p;
      }
    }

    // P as A-fragment (wave-private LDS round trip; DS in-order per wave).
    v8h plo = *(const v8h*)(&ldsP[wave][n * 32 + 16 * g]);
    v8h phi = *(const v8h*)(&ldsP[wave][n * 32 + 16 * g + 8]);
    v16h pf;
#pragma unroll
    for (int i = 0; i < 8; ++i) { pf[i] = plo[i]; pf[8 + i] = phi[i]; }

    // O += P(16x32) @ V(32x256): same A across the chain -> reuse_a hint.
#pragma unroll
    for (int c = 0; c < 16; ++c) {
      v16h b = *(const v16h*)(&ldsV[buf][(16 * c + n) * 32 + 16 * g]);
      oacc[c] = (c == 0) ? wmma_f16(pf, b, oacc[c])
                         : wmma_f16_ra(pf, b, oacc[c]);
    }

    wait_asynccnt0();      // next tile fully in LDS (this wave's DMAs)
    __syncthreads();       // ... and everyone else's
  }

#pragma unroll
  for (int c = 0; c < 16; ++c)
#pragma unroll
    for (int r = 0; r < 8; ++r)
      out[(size_t)(qrow0 + 8 * g + r) * HID + 16 * c + n] = oacc[c][r];
}

// ---------------------------------------------------------------------------
extern "C" void kernel_launch(void* const* d_in, const int* in_sizes, int n_in,
                              void* d_out, int out_size, void* d_ws, size_t ws_size,
                              hipStream_t stream) {
  (void)in_sizes; (void)n_in; (void)out_size; (void)ws_size;
  const float* query = (const float*)d_in[0];
  const float* key   = (const float*)d_in[1];
  const float* value = (const float*)d_in[2];
  const float* Wq    = (const float*)d_in[3];
  const float* bq    = (const float*)d_in[4];
  const float* Wk    = (const float*)d_in[5];
  const float* bk    = (const float*)d_in[6];
  const float* Wv    = (const float*)d_in[7];
  const float* bv    = (const float*)d_in[8];

  _Float16* Qh = (_Float16*)d_ws;                 // 4 MiB
  _Float16* Kh = Qh + (size_t)N_TOK * HID;        // 4 MiB
  _Float16* Vt = Kh + (size_t)N_TOK * HID;        // 4 MiB, [HID][N_TOK]

  dim3 blk(128);
  dim3 grd(N_TOK / 64);   // 4 waves x 16 q-rows per block
  proj_kernel<<<grd, blk, 0, stream>>>(query, Wq, bq, Qh, 0);
  proj_kernel<<<grd, blk, 0, stream>>>(key,   Wk, bk, Kh, 0);
  proj_kernel<<<grd, blk, 0, stream>>>(value, Wv, bv, Vt, 1);
  attn_kernel<<<grd, blk, 0, stream>>>(Qh, Kh, Vt, (float*)d_out);
}
